// Decoder_75101798137980
// MI455X (gfx1250) — matrix operand order repaired and checked by reading, not compiled, not run on hardware
//
#include <hip/hip_runtime.h>
#include <hip/hip_bf16.h>
#include <math.h>

typedef __attribute__((ext_vector_type(16))) __bf16 v16bf;
typedef __attribute__((ext_vector_type(8)))  __bf16 v8bf;
typedef __attribute__((ext_vector_type(8)))  float  v8f;

#define DEC_B   64
#define DEC_T   1024
#define DEC_KVS 128
#define DEC_H   512
#define DEC_E   256
#define DEC_V   30
#define DEC_L   600
#define X1W     896   // E + KVS + H
#define X2W     640   // H + KVS

__device__ __forceinline__ float sigmf_(float x){ return 1.0f/(1.0f + __expf(-x)); }

// ---------------- prep kernels ----------------
__global__ void cvt_bf16_kernel(const float* __restrict__ src, __bf16* __restrict__ dst, int n){
  int i = blockIdx.x*blockDim.x + threadIdx.x;
  if (i < n) dst[i] = (__bf16)src[i];
}
__global__ void zero_f32_kernel(float* __restrict__ p, int n){
  int i = blockIdx.x*blockDim.x + threadIdx.x;
  if (i < n) p[i] = 0.0f;
}
__global__ void zero_bf16_kernel(__bf16* __restrict__ p, int n){
  int i = blockIdx.x*blockDim.x + threadIdx.x;
  if (i < n) p[i] = (__bf16)0.0f;
}
// X1[0][b] = [ bf16(emb[0,:]) | zeros(ctx) | zeros(h1) ]   (tok0 == 0)
__global__ void init_x1_kernel(__bf16* __restrict__ x1, const float* __restrict__ emb){
  int b = blockIdx.x, i = threadIdx.x;                 // blockDim = 896
  __bf16 v = (i < DEC_E) ? (__bf16)emb[i] : (__bf16)0.0f;
  x1[(size_t)b*X1W + i] = v;
}

// ---------------- fused LSTM-cell GEMM (bf16 WMMA, f32 accum) ----------------
// gates[64, 4*Hout] = X[:,0:K1] @ Wi^T + X[:,K1:K1+K2] @ Wh^T  (+biases), then
// c' = sig(f)*c + sig(i)*tanh(g);  h' = sig(o)*tanh(c')
// Grid: Hout/16 WGs. Block: 128 (4 waves). Wave w = M-tile w (rows 16w..16w+15),
// each wave computes the 4 gate tiles for this WG's 16 hidden columns.
__global__ __launch_bounds__(128) void lstm_wmma_kernel(
    const __bf16* __restrict__ X, int Krow,
    const __bf16* __restrict__ Wi, int K1,          // [4*Hout, K1] row-major
    const __bf16* __restrict__ Wh, int K2,          // [4*Hout, K2] row-major
    const float*  __restrict__ bih, const float* __restrict__ bhh,
    float* __restrict__ c, int Hout,                // [64, Hout] in-place
    __bf16* __restrict__ hA, int sA,                // h' bf16 dest A (+col base applied by host)
    __bf16* __restrict__ hB, int sB,                // optional second bf16 dest
    float*  __restrict__ hF)                        // optional f32 dest, stride Hout
{
  const int lane = threadIdx.x & 31;
  const int wave = threadIdx.x >> 5;        // M-tile
  const int lm   = lane & 15;
  const int half = lane >> 4;               // 0 or 1
  const int col  = blockIdx.x*16 + lm;      // hidden column
  const int klo  = half*16;                 // A-operand sub-chunk
  const int kbo  = half*16;                 // B-operand sub-chunk

  const __bf16* Arow = X + (size_t)(wave*16 + lm)*Krow;

  v8f acc[4];
  #pragma unroll
  for (int g = 0; g < 4; ++g)
    #pragma unroll
    for (int r = 0; r < 8; ++r) acc[g][r] = 0.0f;

  const __bf16* Wr1[4];
  const __bf16* Wr2[4];
  #pragma unroll
  for (int g = 0; g < 4; ++g){
    Wr1[g] = Wi + (size_t)(g*Hout + col)*K1;
    Wr2[g] = Wh + (size_t)(g*Hout + col)*K2;
  }

  // K segment 1: input weights
  for (int k = 0; k < K1; k += 32){
    v8bf a0 = *(const v8bf*)(Arow + k + klo);
    v8bf a1 = *(const v8bf*)(Arow + k + klo + 8);
    v16bf a = __builtin_shufflevector(a0, a1, 0,1,2,3,4,5,6,7,8,9,10,11,12,13,14,15);
    #pragma unroll
    for (int g = 0; g < 4; ++g){
      v16bf b = *(const v16bf*)(Wr1[g] + k + kbo);
      acc[g] = __builtin_amdgcn_wmma_f32_16x16x32_bf16(false, a, false, b,
                                                       (short)0, acc[g], false, false);
    }
  }
  // K segment 2: recurrent weights (A continues at column K1 of X)
  for (int k = 0; k < K2; k += 32){
    v8bf a0 = *(const v8bf*)(Arow + K1 + k + klo);
    v8bf a1 = *(const v8bf*)(Arow + K1 + k + klo + 8);
    v16bf a = __builtin_shufflevector(a0, a1, 0,1,2,3,4,5,6,7,8,9,10,11,12,13,14,15);
    #pragma unroll
    for (int g = 0; g < 4; ++g){
      v16bf b = *(const v16bf*)(Wr2[g] + k + kbo);
      acc[g] = __builtin_amdgcn_wmma_f32_16x16x32_bf16(false, a, false, b,
                                                       (short)0, acc[g], false, false);
    }
  }

  float bias[4];
  #pragma unroll
  for (int g = 0; g < 4; ++g) bias[g] = bih[g*Hout + col] + bhh[g*Hout + col];

  const int mbase = wave*16 + half*8;       // C/D layout: VGPR r -> row r + 8*half
  #pragma unroll
  for (int r = 0; r < 8; ++r){
    int m = mbase + r;
    float iv = acc[0][r] + bias[0];
    float fv = acc[1][r] + bias[1];
    float gv = acc[2][r] + bias[2];
    float ov = acc[3][r] + bias[3];
    size_t ci = (size_t)m*Hout + col;
    float cn = sigmf_(fv)*c[ci] + sigmf_(iv)*tanhf(gv);
    c[ci] = cn;
    float hn = sigmf_(ov)*tanhf(cn);
    __bf16 hb = (__bf16)hn;
    hA[(size_t)m*sA + col] = hb;
    if (hB) hB[(size_t)m*sB + col] = hb;
    if (hF) hF[ci] = hn;
  }
}

// ---------------- attention + projection + argmax + feedback ----------------
// One WG per batch row. energy -> masked softmax -> ctx -> logits -> argmax,
// writes Y/attn outputs and the next step's [emb(tok)|ctx] bf16 into X1_next.
__global__ __launch_bounds__(256) void attn_step_kernel(
    const float* __restrict__ key, const float* __restrict__ value,
    const int*   __restrict__ lens, const float* __restrict__ h2,
    const float* __restrict__ emb,  const float* __restrict__ Wp,
    const float* __restrict__ bp,
    __bf16* __restrict__ x1n, float* __restrict__ Y, float* __restrict__ A, int step)
{
  __shared__ float sh_h2[DEC_KVS];
  __shared__ float sh_e[DEC_T];
  __shared__ float red[256];
  __shared__ float sh_ctx[DEC_KVS];
  __shared__ float sh_logit[DEC_V];
  __shared__ int   sh_tok;

  const int b = blockIdx.x, tid = threadIdx.x;
  if (tid < DEC_KVS) sh_h2[tid] = h2[b*DEC_KVS + tid];
  __syncthreads();

  const int L = lens[b];
  float pmax = -INFINITY;
  #pragma unroll
  for (int tt = 0; tt < 4; ++tt){
    int t = tid + tt*256;
    const float4* kr = (const float4*)(key + ((size_t)b*DEC_T + t)*DEC_KVS);
    float s = 0.0f;
    #pragma unroll 8
    for (int j = 0; j < DEC_KVS/4; ++j){
      float4 kv = kr[j];
      s += kv.x*sh_h2[4*j] + kv.y*sh_h2[4*j+1] + kv.z*sh_h2[4*j+2] + kv.w*sh_h2[4*j+3];
    }
    float e = (t >= L) ? -INFINITY : s;
    sh_e[t] = e;
    pmax = fmaxf(pmax, e);
  }
  red[tid] = pmax; __syncthreads();
  for (int off = 128; off > 0; off >>= 1){
    if (tid < off) red[tid] = fmaxf(red[tid], red[tid+off]);
    __syncthreads();
  }
  const float mx = red[0]; __syncthreads();

  float ex4[4]; float psum = 0.0f;
  #pragma unroll
  for (int tt = 0; tt < 4; ++tt){
    int t = tid + tt*256;
    float ex = __expf(sh_e[t] - mx);      // -inf -> 0
    ex4[tt] = ex; psum += ex;
  }
  red[tid] = psum; __syncthreads();
  for (int off = 128; off > 0; off >>= 1){
    if (tid < off) red[tid] += red[tid+off];
    __syncthreads();
  }
  const float inv = 1.0f/red[0]; __syncthreads();
  #pragma unroll
  for (int tt = 0; tt < 4; ++tt){
    int t = tid + tt*256;
    float a = ex4[tt]*inv;
    sh_e[t] = a;
    A[((size_t)b*DEC_L + step)*DEC_T + t] = a;
  }
  __syncthreads();

  // ctx[k] = sum_t attn[t] * value[b,t,k]; two half-T partial groups
  const int kk = tid & (DEC_KVS-1), grp = tid >> 7;
  const float* vb = value + (size_t)b*DEC_T*DEC_KVS;
  float cp = 0.0f;
  for (int t = grp*(DEC_T/2); t < (grp+1)*(DEC_T/2); ++t)
    cp += sh_e[t]*vb[(size_t)t*DEC_KVS + kk];
  red[tid] = cp; __syncthreads();
  if (tid < DEC_KVS) sh_ctx[tid] = red[tid] + red[tid+128];
  __syncthreads();

  if (tid < DEC_V){
    const float* wr = Wp + tid*(2*DEC_KVS);
    float s = bp[tid];
    for (int j = 0; j < DEC_KVS; ++j) s += wr[j]*sh_h2[j];
    for (int j = 0; j < DEC_KVS; ++j) s += wr[DEC_KVS + j]*sh_ctx[j];
    sh_logit[tid] = s;
    Y[((size_t)b*DEC_L + step)*DEC_V + tid] = s;
  }
  __syncthreads();
  if (tid == 0){
    float best = sh_logit[0]; int bi = 0;
    for (int v = 1; v < DEC_V; ++v) if (sh_logit[v] > best){ best = sh_logit[v]; bi = v; }
    sh_tok = bi;
  }
  __syncthreads();
  if (tid < DEC_E)   x1n[(size_t)b*X1W + tid]         = (__bf16)emb[sh_tok*DEC_E + tid];
  if (tid < DEC_KVS) x1n[(size_t)b*X1W + DEC_E + tid] = (__bf16)sh_ctx[tid];
}

// ---------------- host ----------------
extern "C" void kernel_launch(void* const* d_in, const int* in_sizes, int n_in,
                              void* d_out, int out_size, void* d_ws, size_t ws_size,
                              hipStream_t stream){
  (void)in_sizes; (void)n_in; (void)out_size; (void)ws_size;
  const float* key   = (const float*)d_in[0];
  const float* value = (const float*)d_in[1];
  const int*   lens  = (const int*)  d_in[2];
  const float* emb   = (const float*)d_in[3];
  const float* Wih1  = (const float*)d_in[4];
  const float* Whh1  = (const float*)d_in[5];
  const float* bih1  = (const float*)d_in[6];
  const float* bhh1  = (const float*)d_in[7];
  const float* Wih2  = (const float*)d_in[8];
  const float* Whh2  = (const float*)d_in[9];
  const float* bih2  = (const float*)d_in[10];
  const float* bhh2  = (const float*)d_in[11];
  const float* Wproj = (const float*)d_in[12];
  const float* bproj = (const float*)d_in[13];
  // d_in[14] = max_len (fixed at 600 in the reference setup)

  char* ws = (char*)d_ws; size_t off = 0;
  auto carve = [&](size_t bytes)->char*{
    char* p = ws + off; off += (bytes + 255) & ~(size_t)255; return p;
  };
  const int nW1i = 4*DEC_H*(DEC_E+DEC_KVS);   // 2048*384
  const int nW1h = 4*DEC_H*DEC_H;             // 2048*512
  const int nW2i = 4*DEC_KVS*DEC_H;           // 512*512
  const int nW2h = 4*DEC_KVS*DEC_KVS;         // 512*128
  __bf16* W1i = (__bf16*)carve((size_t)nW1i*2);
  __bf16* W1h = (__bf16*)carve((size_t)nW1h*2);
  __bf16* W2i = (__bf16*)carve((size_t)nW2i*2);
  __bf16* W2h = (__bf16*)carve((size_t)nW2h*2);
  __bf16* X1a = (__bf16*)carve((size_t)DEC_B*X1W*2);
  __bf16* X1b = (__bf16*)carve((size_t)DEC_B*X1W*2);
  __bf16* X2a = (__bf16*)carve((size_t)DEC_B*X2W*2);
  __bf16* X2b = (__bf16*)carve((size_t)DEC_B*X2W*2);
  float*  c1  = (float*) carve((size_t)DEC_B*DEC_H*4);
  float*  c2  = (float*) carve((size_t)DEC_B*DEC_KVS*4);
  float*  h2f = (float*) carve((size_t)DEC_B*DEC_KVS*4);

  float* Y = (float*)d_out;
  float* A = Y + (size_t)DEC_B*DEC_L*DEC_V;

  auto cvt = [&](const float* s, __bf16* d, int n){
    cvt_bf16_kernel<<<(n+255)/256, 256, 0, stream>>>(s, d, n);
  };
  cvt(Wih1, W1i, nW1i); cvt(Whh1, W1h, nW1h);
  cvt(Wih2, W2i, nW2i); cvt(Whh2, W2h, nW2h);
  zero_f32_kernel<<<(DEC_B*DEC_H+255)/256, 256, 0, stream>>>(c1, DEC_B*DEC_H);
  zero_f32_kernel<<<(DEC_B*DEC_KVS+255)/256, 256, 0, stream>>>(c2, DEC_B*DEC_KVS);
  zero_bf16_kernel<<<(DEC_B*X2W+255)/256, 256, 0, stream>>>(X2a, DEC_B*X2W);
  init_x1_kernel<<<DEC_B, X1W, 0, stream>>>(X1a, emb);

  for (int s = 0; s < DEC_L; ++s){
    const bool even = (s & 1) == 0;
    __bf16* X1c = even ? X1a : X1b;  __bf16* X1n = even ? X1b : X1a;
    __bf16* X2c = even ? X2a : X2b;  __bf16* X2n = even ? X2b : X2a;

    // LSTM1: X=[emb|ctx|h1] (K=384+512), gates -> h1',c1'
    lstm_wmma_kernel<<<DEC_H/16, 128, 0, stream>>>(
        X1c, X1W, W1i, DEC_E+DEC_KVS, W1h, DEC_H, bih1, bhh1, c1, DEC_H,
        X1n + (DEC_E+DEC_KVS), X1W,   // h1' -> next X1
        X2c + 0, X2W,                 // h1' -> current X2 (read by LSTM2 this step)
        nullptr);
    // LSTM2: X=[h1'|h2] (K=512+128), gates -> h2',c2'
    lstm_wmma_kernel<<<DEC_KVS/16, 128, 0, stream>>>(
        X2c, X2W, W2i, DEC_H, W2h, DEC_KVS, bih2, bhh2, c2, DEC_KVS,
        X2n + DEC_H, X2W,             // h2' -> next X2
        nullptr, 0,
        h2f);                         // h2' f32 for attention/logits
    // attention + proj + argmax; writes Y/attn and next-step [emb|ctx]
    attn_step_kernel<<<DEC_B, 256, 0, stream>>>(
        key, value, lens, h2f, emb, Wproj, bproj, X1n, Y, A, s);
  }
}
